// GroupQueryAttention_43954695307310
// MI455X (gfx1250) — matrix operand order repaired and checked by reading, not compiled, not run on hardware
//
#include <hip/hip_runtime.h>

// ---------------------------------------------------------------------------
// Problem constants (from the reference)
// ---------------------------------------------------------------------------
#define B_ 2
#define L_ 128
#define E_ 1024
#define H_ 16
#define G_ 8
#define D_ 64
#define R_ 2

typedef __attribute__((ext_vector_type(16))) __bf16 bf16x16;
typedef __attribute__((ext_vector_type(8)))  __bf16 bf16x8;
typedef __attribute__((ext_vector_type(8)))  float  f32x8;

__device__ __forceinline__ f32x8 f32x8_zero() {
  f32x8 v = {0.f, 0.f, 0.f, 0.f, 0.f, 0.f, 0.f, 0.f};
  return v;
}

__device__ __forceinline__ bf16x16 frag_cat(const __bf16* p0, const __bf16* p1) {
  bf16x8 lo = *(const bf16x8*)p0;
  bf16x8 hi = *(const bf16x8*)p1;
  return __builtin_shufflevector(lo, hi, 0, 1, 2, 3, 4, 5, 6, 7,
                                 8, 9, 10, 11, 12, 13, 14, 15);
}

// A fragment, 16x32 (MxK) bf16. ISA layout: lane<16 holds row (lane&15),
// K = {0..7} U {16..23}; lane>=16 holds K = {8..15} U {24..31}.
__device__ __forceinline__ bf16x16 load_a_frag(const __bf16* base, int lda) {
  int lane = threadIdx.x & 31;
  const __bf16* p = base + (lane & 15) * lda + ((lane < 16) ? 0 : 8);
  return frag_cat(p, p + 16);
}

// B fragment, 32x16 (KxN) bf16 from K-major LDS: lane holds column (lane&15),
// K = {0..7} U {16..23} (lanes 0-15) or {8..15} U {24..31} (lanes 16-31).
__device__ __forceinline__ bf16x16 load_b_frag(const __bf16* base, int ldw) {
  int lane = threadIdx.x & 31;
  const __bf16* p = base + (lane & 15) * ldw + ((lane < 16) ? 0 : 8);
  return frag_cat(p, p + 16);
}

__device__ __forceinline__ f32x8 wmma_bf16(bf16x16 a, bf16x16 b, f32x8 c) {
  return __builtin_amdgcn_wmma_f32_16x16x32_bf16(false, a, false, b,
                                                 (short)0, c, false, false);
}

// ---------------------------------------------------------------------------
// Pass 0: transpose-convert weights f32 [K,N] -> bf16 [N,K] (K-major), so the
// hot loops stage B tiles with contiguous b128 global loads + b128 LDS stores.
// Runs once; weights then live in the 192MB L2.
// ---------------------------------------------------------------------------
__global__ __launch_bounds__(256) void cvt_t_kernel(const float* __restrict__ src,
                                                    __bf16* __restrict__ dst,
                                                    int K, int N) {
  __shared__ __bf16 tile[32][33];
  const int k0 = blockIdx.x * 32;
  const int n0 = blockIdx.y * 32;
  const int tx = threadIdx.x & 31;
  const int ty = threadIdx.x >> 5;  // 0..7
  for (int j = 0; j < 4; ++j)
    tile[ty + 8 * j][tx] =
        (__bf16)src[(size_t)(k0 + ty + 8 * j) * N + n0 + tx];
  __syncthreads();
  for (int j = 0; j < 4; ++j)
    dst[(size_t)(n0 + ty + 8 * j) * K + k0 + tx] = tile[tx][ty + 8 * j];
}

// ---------------------------------------------------------------------------
// Generic WMMA GEMM: out[M,N] = Af32[M,K] @ Wt(bf16,[N,K]) + bias[N]
// Block tile 64M x 128N, 4 waves, each wave 32M x 64N (8 WMMA accumulators).
// Direct staging (load -> cvt -> b128 LDS store) — no register pipeline, the
// round-3 version spilled the pipelined slab to scratch. launch_bounds(.,1)
// lifts the VGPR occupancy cap (these GEMMs are <2% of total work).
// M % 64 == 0, N % 128 == 0, K % 32 == 0.
// ---------------------------------------------------------------------------
__global__ __launch_bounds__(128, 1) void gemm_bf16_kernel(
    const float* __restrict__ A, const __bf16* __restrict__ Wt,
    const float* __restrict__ bias, float* __restrict__ out,
    int M, int N, int K) {
  __shared__ __align__(16) __bf16 sA[64 * 40];    // [row][k], pad to 40
  __shared__ __align__(16) __bf16 sW[128 * 40];   // [n][k], K-major
  const int LDA = 40, LDW = 40;
  const int m0 = blockIdx.x * 64;
  const int n0 = blockIdx.y * 128;
  const int t = threadIdx.x;
  const int lane = t & 31;
  const int wave = t >> 5;
  const int wm = wave >> 1, wn = wave & 1;

  f32x8 acc[2][4];
  for (int mt = 0; mt < 2; ++mt)
    for (int nt = 0; nt < 4; ++nt) acc[mt][nt] = f32x8_zero();

  const int ar = t >> 1;            // A row 0..63
  const int acb = (t & 1) * 16;     // A k-half
  const float* arow = A + (size_t)(m0 + ar) * K + acb;
  const __bf16* wrow = Wt + (size_t)(n0 + t) * K;

  for (int kk = 0; kk < K; kk += 32) {
    __syncthreads();
    {  // stage A: 64 rows x 32 k, f32 -> bf16 (field-wise, no ptr casts)
      float4 x[4];
      const float4* src = (const float4*)(arow + kk);
#pragma unroll
      for (int q = 0; q < 4; ++q) x[q] = src[q];
      __bf16* d = sA + ar * LDA + acb;
#pragma unroll
      for (int q = 0; q < 4; ++q) {
        d[4 * q + 0] = (__bf16)x[q].x;
        d[4 * q + 1] = (__bf16)x[q].y;
        d[4 * q + 2] = (__bf16)x[q].z;
        d[4 * q + 3] = (__bf16)x[q].w;
      }
    }
    {  // stage W: one n-row per thread, 32 contiguous k -> 4 b128 stores
      uint4 wb[4];
      const uint4* s4 = (const uint4*)(wrow + kk);
#pragma unroll
      for (int q = 0; q < 4; ++q) wb[q] = s4[q];
      __bf16* dw = sW + t * LDW;
#pragma unroll
      for (int q = 0; q < 4; ++q) *(uint4*)(dw + 8 * q) = wb[q];
      if (kk + 32 < K) __builtin_prefetch(wrow + kk + 32, 0, 1);
    }
    __syncthreads();
#pragma unroll
    for (int mt = 0; mt < 2; ++mt) {
      bf16x16 af = load_a_frag(sA + (wm * 32 + mt * 16) * LDA, LDA);
#pragma unroll
      for (int nt = 0; nt < 4; ++nt) {
        bf16x16 bfr = load_b_frag(sW + (wn * 64 + nt * 16) * LDW, LDW);
        acc[mt][nt] = wmma_bf16(af, bfr, acc[mt][nt]);
      }
    }
  }
  // Epilogue: C layout — lane<16: M=vgpr, N=lane; lane>=16: M=vgpr+8, N=lane-16
  const int nl = lane & 15;
  const int mh = (lane < 16) ? 0 : 8;
  for (int mt = 0; mt < 2; ++mt)
    for (int nt = 0; nt < 4; ++nt) {
      int col = n0 + wn * 64 + nt * 16 + nl;
      float bvv = bias[col];
      for (int r = 0; r < 8; ++r) {
        int row = m0 + wm * 32 + mt * 16 + r + mh;
        out[(size_t)row * N + col] = acc[mt][nt][r] + bvv;
      }
    }
}

// ---------------------------------------------------------------------------
// Fused: LayerNorm(rpe rows) -> [rn@Wrk | rn@Wrv] (bf16 WMMA) -> scores +
// rpe_v(bf16). One workgroup per (b, i): the full 128x1024 normalized bf16
// A-tile lives in LDS (264KB — possible only with CDNA5's 320KB/WGP) and is
// reused across all 8 KV groups; rpe_k never touches memory (the D=64 q-dot
// is reduced out of the WMMA accumulators with cross-lane shuffles).
// 256 threads = 8 waves (2/SIMD, declared via launch_bounds), wave grid
// 4M x 2N, wave tile 32M x 64N (2x4 accumulators -> 12 ds_load_b128 per
// 8 WMMAs). Weight K-slabs (128k x 128n, 32KB) register-pipelined.
// ---------------------------------------------------------------------------
#define RPE_LDA 1032                       // 1024 + 8 pad (4-bank row skew)
#define RPE_LDW 136                        // 128 + 8 pad
#define RPE_SMEM (128 * RPE_LDA * 2 + 128 * RPE_LDW * 2)  // 299,008 B

__global__ __launch_bounds__(256, 2) void rpe_fused_kernel(
    const float* __restrict__ rpe, const float* __restrict__ lng,
    const float* __restrict__ lnb, const __bf16* __restrict__ Wrkt,
    const __bf16* __restrict__ Wrvt, const float* __restrict__ brk,
    const float* __restrict__ brv, const float* __restrict__ qbuf,
    const float* __restrict__ kbuf, float* __restrict__ scores,
    __bf16* __restrict__ rpev) {
  extern __shared__ __align__(16) unsigned char smem[];
  __bf16* sA = (__bf16*)smem;                         // [128][RPE_LDA]
  __bf16* sW = (__bf16*)(smem + 128 * RPE_LDA * 2);   // [128][RPE_LDW], K-major
  const int LDA = RPE_LDA, LDW = RPE_LDW;

  const int blk = blockIdx.x;  // B*L
  const int i = blk & (L_ - 1);
  const int b = blk >> 7;

  const int t = threadIdx.x;   // 0..255
  const int lane = t & 31;
  const int wave = t >> 5;     // 0..7
  const int wm = wave >> 1;    // 0..3 -> 32 rows each
  const int wn = wave & 1;     // 0: rpe_k cols, 1: rpe_v cols

  // ---- LayerNorm 128 rpe rows (b, i, row, :) into sA as bf16 ----
  for (int row = wave; row < L_; row += 8) {
    const float* x = rpe + ((size_t)((b * L_ + i) * L_) + row) * E_;
    float4 xv[8];
    float s = 0.f, ss = 0.f;
#pragma unroll
    for (int c = 0; c < 8; ++c) {
      float4 q4 = ((const float4*)x)[lane + c * 32];
      xv[c] = q4;
      s += q4.x + q4.y + q4.z + q4.w;
      ss += q4.x * q4.x + q4.y * q4.y + q4.z * q4.z + q4.w * q4.w;
    }
    for (int m = 16; m >= 1; m >>= 1) {
      s += __shfl_xor(s, m, 32);
      ss += __shfl_xor(ss, m, 32);
    }
    float mu = s * (1.f / E_);
    float var = ss * (1.f / E_) - mu * mu;
    float rs = __frsqrt_rn(var + 1e-5f);
    __bf16* drow = sA + row * LDA;
#pragma unroll
    for (int c = 0; c < 8; ++c) {
      int col = (lane + c * 32) * 4;
      float4 g4 = ((const float4*)lng)[lane + c * 32];
      float4 b4 = ((const float4*)lnb)[lane + c * 32];
      drow[col + 0] = (__bf16)((xv[c].x - mu) * rs * g4.x + b4.x);
      drow[col + 1] = (__bf16)((xv[c].y - mu) * rs * g4.y + b4.y);
      drow[col + 2] = (__bf16)((xv[c].z - mu) * rs * g4.z + b4.z);
      drow[col + 3] = (__bf16)((xv[c].w - mu) * rs * g4.w + b4.w);
    }
  }

  const float scale = 0.125f;  // 1/sqrt(D)
  // Per-thread weight staging: n-row (t>>1), 64 contiguous k ((t&1)*64)
  const int sn = t >> 1;
  const int skh = (t & 1) * 64;
  const __bf16* wsrc_base = (sn < 64) ? Wrkt : Wrvt;
  const int srow = (sn & 63);

  for (int g = 0; g < G_; ++g) {
    f32x8 acc[2][4];
#pragma unroll
    for (int mt = 0; mt < 2; ++mt)
#pragma unroll
      for (int nt = 0; nt < 4; ++nt) acc[mt][nt] = f32x8_zero();

    const __bf16* wrow = wsrc_base + (size_t)(g * 64 + srow) * E_ + skh;
    uint4 wreg[8];
    {  // prologue: fetch slab kk0 = 0 into registers
      const uint4* s4 = (const uint4*)(wrow);
#pragma unroll
      for (int q = 0; q < 8; ++q) wreg[q] = s4[q];
    }

    for (int kk0 = 0; kk0 < E_; kk0 += 128) {
      __syncthreads();  // previous slab's WMMAs done (also guards LN on g=0)
      {  // commit staged registers to LDS (b128 stores, K-major)
        __bf16* dstp = sW + sn * LDW + skh;
#pragma unroll
        for (int q = 0; q < 8; ++q) *(uint4*)(dstp + 8 * q) = wreg[q];
      }
      __syncthreads();
      if (kk0 + 128 < E_) {  // pipeline: fetch next slab during compute
        const uint4* s4 = (const uint4*)(wrow + kk0 + 128);
#pragma unroll
        for (int q = 0; q < 8; ++q) wreg[q] = s4[q];
      }
#pragma unroll
      for (int kloc = 0; kloc < 128; kloc += 32) {
#pragma unroll
        for (int mt = 0; mt < 2; ++mt) {
          bf16x16 af =
              load_a_frag(sA + (wm * 32 + mt * 16) * LDA + kk0 + kloc, LDA);
#pragma unroll
          for (int nt = 0; nt < 4; ++nt) {
            bf16x16 bfr =
                load_b_frag(sW + (wn * 64 + nt * 16) * LDW + kloc, LDW);
            acc[mt][nt] = wmma_bf16(af, bfr, acc[mt][nt]);
          }
        }
      }
    }

    const int nl = lane & 15;
    const int mh = (lane < 16) ? 0 : 8;
    if (wn == 0) {
      // accumulators = rpe_k tile [32 j][64 d]: add brk, q-dot, emit scores
#pragma unroll
      for (int nt = 0; nt < 4; ++nt) {
        float bvv = brk[g * 64 + nt * 16 + nl];
#pragma unroll
        for (int mt = 0; mt < 2; ++mt)
#pragma unroll
          for (int r = 0; r < 8; ++r) acc[mt][nt][r] += bvv;
      }
      for (int rq = 0; rq < R_; ++rq) {
        const float* qrow =
            qbuf + ((size_t)(b * L_ + i) * H_ + (g * R_ + rq)) * D_;
        const float* krow = kbuf + ((size_t)(b * L_ + i) * G_ + g) * D_;
        float bp = qrow[lane] * krow[lane] + qrow[lane + 32] * krow[lane + 32];
        for (int m = 16; m >= 1; m >>= 1) bp += __shfl_xor(bp, m, 32);
        float qf[4];
#pragma unroll
        for (int nt = 0; nt < 4; ++nt) qf[nt] = qrow[nt * 16 + nl];
        size_t sb = ((((size_t)(b * G_ + g) * R_) + rq) * L_ + i) * L_;
        for (int mt = 0; mt < 2; ++mt) {
          float pr[8];
#pragma unroll
          for (int r = 0; r < 8; ++r) {
            float x = 0.f;
#pragma unroll
            for (int nt = 0; nt < 4; ++nt) x += acc[mt][nt][r] * qf[nt];
            for (int m = 8; m >= 1; m >>= 1) x += __shfl_xor(x, m, 32);
            pr[r] = x;
          }
          if (nl == 0) {
            for (int r = 0; r < 8; ++r) {
              int j = wm * 32 + mt * 16 + r + mh;
              scores[sb + j] = (bp + pr[r]) * scale;
            }
          }
        }
      }
    } else {
      // accumulators = rpe_v tile: add brv, store bf16 [B,G,L,L,D]
#pragma unroll
      for (int nt = 0; nt < 4; ++nt) {
        int d = nt * 16 + nl;
        float bvv = brv[g * 64 + d];
#pragma unroll
        for (int mt = 0; mt < 2; ++mt)
#pragma unroll
          for (int r = 0; r < 8; ++r) {
            int j = wm * 32 + mt * 16 + r + mh;
            size_t idx =
                ((((size_t)(b * G_ + g) * L_) + i) * L_ + j) * D_ + d;
            rpev[idx] = (__bf16)(acc[mt][nt][r] + bvv);
          }
      }
    }
    if (g + 1 < G_)
      __builtin_prefetch(wsrc_base + (size_t)((g + 1) * 64 + srow) * E_ + skh,
                         0, 1);
  }
}

// ---------------------------------------------------------------------------
// Global softmax stats per (b,g,r): max and sum(exp) over the flattened L*L
// axis (faithful to the reference's reshape-then-softmax).
// ---------------------------------------------------------------------------
__global__ __launch_bounds__(256) void softmax_stats_kernel(
    const float* __restrict__ scores, float* __restrict__ stats) {
  __shared__ float red[256];
  const int gidx = blockIdx.x;  // B*G*R = 32
  const float* s = scores + (size_t)gidx * (L_ * L_);
  const int t = threadIdx.x;
  float mx = -1e30f;
  for (int idx = t; idx < L_ * L_; idx += 256) mx = fmaxf(mx, s[idx]);
  red[t] = mx;
  __syncthreads();
  for (int off = 128; off > 0; off >>= 1) {
    if (t < off) red[t] = fmaxf(red[t], red[t + off]);
    __syncthreads();
  }
  mx = red[0];
  __syncthreads();
  float sum = 0.f;
  for (int idx = t; idx < L_ * L_; idx += 256) sum += __expf(s[idx] - mx);
  red[t] = sum;
  __syncthreads();
  for (int off = 128; off > 0; off >>= 1) {
    if (t < off) red[t] += red[t + off];
    __syncthreads();
  }
  if (t == 0) {
    stats[gidx * 2] = mx;
    stats[gidx * 2 + 1] = red[0];
  }
}

// ---------------------------------------------------------------------------
// Attention output: qv[b,i,h,d] = wsum * v[b,i,g,d] + sum_j attn * rpe_v
// (memory-bound bf16 stream of rpe_v). One 2-wave block per (b,g,r,i).
// ---------------------------------------------------------------------------
__global__ __launch_bounds__(64) void attend_kernel(
    const float* __restrict__ scores, const float* __restrict__ stats,
    const __bf16* __restrict__ rpev, const float* __restrict__ vbuf,
    float* __restrict__ qv) {
  const int blk = blockIdx.x;  // B*G*R*L = 4096
  const int i = blk & (L_ - 1);
  const int r = (blk >> 7) & 1;
  const int g = (blk >> 8) & 7;
  const int b = blk >> 11;
  const int d = threadIdx.x;  // 0..63
  const int gidx = (b * G_ + g) * R_ + r;
  const float mx = stats[gidx * 2];
  const float invZ = 1.f / stats[gidx * 2 + 1];
  const float* s = scores + ((size_t)gidx * L_ + i) * L_;
  const __bf16* pv = rpev + (((size_t)(b * G_ + g) * L_ + i) * L_) * D_ + d;
  float acc = 0.f, wsum = 0.f;
  for (int j = 0; j < L_; ++j) {
    float w = __expf(s[j] - mx) * invZ;
    wsum += w;
    acc += w * (float)pv[(size_t)j * D_];
  }
  float vv = vbuf[((size_t)(b * L_ + i) * G_ + g) * D_ + d];
  qv[((size_t)(b * L_ + i) * H_ + (g * R_ + r)) * D_ + d] = wsum * vv + acc;
}

// ---------------------------------------------------------------------------
// Host orchestration. Workspace layout (bytes from d_ws, ~46 MB total):
//   [0,2M)  WqT bf16 [N,K]  [2M,3M) WkT  [3M,4M) WvT
//   [4M,5M) WrkT            [5M,6M) WrvT [6M,8M) WoT
//   [8M,9M) q f32 [B,L,H,D] [9M,9.5M) k f32 [B,L,G,D] [9.5M,10M) v f32
//   [10M,12M) scores f32 [B,G,R,L,L]    [12M,+256B) stats
//   [12M+64K,+1M) qv f32 [B,L,E]        [14M,46M) rpe_v bf16 [B,G,L,L,D]
// ---------------------------------------------------------------------------
extern "C" void kernel_launch(void* const* d_in, const int* in_sizes, int n_in,
                              void* d_out, int out_size, void* d_ws,
                              size_t ws_size, hipStream_t stream) {
  (void)in_sizes; (void)n_in; (void)out_size; (void)ws_size;
  const float* Qi  = (const float*)d_in[0];
  const float* Ki  = (const float*)d_in[1];
  const float* Vi  = (const float*)d_in[2];
  const float* rpe = (const float*)d_in[3];
  const float* lng = (const float*)d_in[4];
  const float* lnb = (const float*)d_in[5];
  const float* Wq  = (const float*)d_in[6];  const float* bq  = (const float*)d_in[7];
  const float* Wk  = (const float*)d_in[8];  const float* bk  = (const float*)d_in[9];
  const float* Wv  = (const float*)d_in[10]; const float* bv  = (const float*)d_in[11];
  const float* Wrk = (const float*)d_in[12]; const float* brk = (const float*)d_in[13];
  const float* Wrv = (const float*)d_in[14]; const float* brv = (const float*)d_in[15];
  const float* Wo  = (const float*)d_in[16]; const float* bo  = (const float*)d_in[17];
  float* out = (float*)d_out;

  unsigned char* w = (unsigned char*)d_ws;
  const size_t MB = 1u << 20;
  __bf16* WqT  = (__bf16*)(w + 0 * MB);
  __bf16* WkT  = (__bf16*)(w + 2 * MB);
  __bf16* WvT  = (__bf16*)(w + 3 * MB);
  __bf16* WrkT = (__bf16*)(w + 4 * MB);
  __bf16* WrvT = (__bf16*)(w + 5 * MB);
  __bf16* WoT  = (__bf16*)(w + 6 * MB);
  float*  qbuf   = (float*)(w + 8 * MB);
  float*  kbuf   = (float*)(w + 9 * MB);
  float*  vbuf   = (float*)(w + 9 * MB + 512 * 1024);
  float*  scores = (float*)(w + 10 * MB);
  float*  stats  = (float*)(w + 12 * MB);
  float*  qv     = (float*)(w + 12 * MB + 64 * 1024);
  __bf16* rpev   = (__bf16*)(w + 14 * MB);

  // Pass 0: transpose-convert all weights to bf16 K-major (once; L2-resident)
  cvt_t_kernel<<<dim3(32, 32), 256, 0, stream>>>(Wq,  WqT,  E_, H_ * D_);
  cvt_t_kernel<<<dim3(32, 16), 256, 0, stream>>>(Wk,  WkT,  E_, G_ * D_);
  cvt_t_kernel<<<dim3(32, 16), 256, 0, stream>>>(Wv,  WvT,  E_, G_ * D_);
  cvt_t_kernel<<<dim3(32, 16), 256, 0, stream>>>(Wrk, WrkT, E_, G_ * D_);
  cvt_t_kernel<<<dim3(32, 16), 256, 0, stream>>>(Wrv, WrvT, E_, G_ * D_);
  cvt_t_kernel<<<dim3(32, 32), 256, 0, stream>>>(Wo,  WoT,  E_, E_);

  // QKV projections (M=B*L=256)
  gemm_bf16_kernel<<<dim3(4, 8), 128, 0, stream>>>(Qi, WqT, bq, qbuf,
                                                   B_ * L_, H_ * D_, E_);
  gemm_bf16_kernel<<<dim3(4, 4), 128, 0, stream>>>(Ki, WkT, bk, kbuf,
                                                   B_ * L_, G_ * D_, E_);
  gemm_bf16_kernel<<<dim3(4, 4), 128, 0, stream>>>(Vi, WvT, bv, vbuf,
                                                   B_ * L_, G_ * D_, E_);

  // Fused LN + RPE dual-GEMM + scores + rpe_v (the 68.7 GFLOP hot loop)
  hipFuncSetAttribute((const void*)rpe_fused_kernel,
                      hipFuncAttributeMaxDynamicSharedMemorySize, RPE_SMEM);
  rpe_fused_kernel<<<B_ * L_, 256, RPE_SMEM, stream>>>(
      rpe, lng, lnb, WrkT, WrvT, brk, brv, qbuf, kbuf, scores, rpev);

  softmax_stats_kernel<<<B_ * G_ * R_, 256, 0, stream>>>(scores, stats);
  attend_kernel<<<B_ * G_ * R_ * L_, 64, 0, stream>>>(scores, stats, rpev,
                                                      vbuf, qv);

  // Output projection
  gemm_bf16_kernel<<<dim3(4, 8), 128, 0, stream>>>(qv, WoT, bo, out,
                                                   B_ * L_, E_, E_);
}
